// SimpleRetention_71262097375687
// MI455X (gfx1250) — matrix operand order repaired and checked by reading, not compiled, not run on hardware
//
#include <hip/hip_runtime.h>
#include <hip/hip_bf16.h>

// ---------------------------------------------------------------------------
// SimpleRetention forward for MI455X (gfx1250), all matmuls via
// v_wmma_f32_16x16x32_bf16 (wave32).
//
// Shapes: B=16, C=2048, EMBED=1024, HEAD=256, KAPPA=0.9
// ---------------------------------------------------------------------------

typedef __bf16 bf16_t;
typedef bf16_t v16bf __attribute__((ext_vector_type(16)));
typedef bf16_t v8bf  __attribute__((ext_vector_type(8)));
typedef float  v8f   __attribute__((ext_vector_type(8)));

#define LN_KAPPA (-0.1053605156578263f)   // logf(0.9)

__device__ __forceinline__ v8f wmma_bf16(v16bf a, v16bf b, v8f c) {
  // 8 args: (neg_a, A, neg_b, B, c_mod, C, reuse_a, reuse_b)
  return __builtin_amdgcn_wmma_f32_16x16x32_bf16(false, a, false, b,
                                                 (short)0, c, false, false);
}

// A fragment (16x32 bf16) from a row-major bf16 row pointer.
// Lane layout: M = lane&15 (both halves). K(e) = e + 8*half (e<8),
//              K(e) = e + 8 + 8*half (e>=8)  -> two contiguous 8-elem chunks.
__device__ __forceinline__ v16bf a_frag_bf16(const bf16_t* row, int k0, int half) {
  v8bf lo = *reinterpret_cast<const v8bf*>(row + k0 + 8 * half);
  v8bf hi = *reinterpret_cast<const v8bf*>(row + k0 + 16 + 8 * half);
  return __builtin_shufflevector(lo, hi, 0,1,2,3,4,5,6,7,8,9,10,11,12,13,14,15);
}

// B fragment (32x16 bf16) from "B^T row-major" (N x K) row pointer.
// Lane layout: N = lane&15, K(e) = e + 8*half (e<8), e + 8 + 8*half (e>=8).
__device__ __forceinline__ v16bf b_frag_bf16(const bf16_t* rowN, int k0, int half) {
  v8bf lo = *reinterpret_cast<const v8bf*>(rowN + k0 + 8 * half);
  v8bf hi = *reinterpret_cast<const v8bf*>(rowN + k0 + 16 + 8 * half);
  return __builtin_shufflevector(lo, hi, 0,1,2,3,4,5,6,7,8,9,10,11,12,13,14,15);
}

// ---------------------------------------------------------------------------
// Kernel 1: projection  Y(32768x256) = X(32768x1024) @ W(256x1024)^T  in bf16.
// One block = 128 rows x 256 cols. 8 waves, wave w owns rows [16w,16w+16).
// W k-slab is staged+converted to bf16 in LDS (it is exactly the N x K
// layout needed for B fragments). Optionally emits row-major Y and/or
// transposed Y^T (HEAD x C per batch) directly from D fragments.
// ---------------------------------------------------------------------------
__global__ __launch_bounds__(256) void proj_kernel(
    const float* __restrict__ X, const float* __restrict__ W,
    bf16_t* __restrict__ Yrow, bf16_t* __restrict__ Ytr) {
  __shared__ bf16_t ldsW[256 * 32];   // 16 KB: W[:, k0:k0+32] in bf16

  const int tid  = threadIdx.x;
  const int wave = tid >> 5, lane = tid & 31;
  const int half = lane >> 4, ln = lane & 15;
  const int r0   = blockIdx.x * 128;

  const float* xrow = X + (size_t)(r0 + 16 * wave + ln) * 1024;

  v8f acc[16];
#pragma unroll
  for (int t = 0; t < 16; ++t) acc[t] = (v8f){0.f,0.f,0.f,0.f,0.f,0.f,0.f,0.f};

  for (int k0 = 0; k0 < 1024; k0 += 32) {
    __syncthreads();
    {  // stage W[tid][k0:k0+32] -> bf16 LDS (row `tid` = output column n)
      const float4* wp = reinterpret_cast<const float4*>(W + (size_t)tid * 1024 + k0);
      bf16_t* dst = ldsW + tid * 32;
#pragma unroll
      for (int j = 0; j < 4; ++j) {
        float4 f0 = wp[2 * j], f1 = wp[2 * j + 1];
        v8bf p;
        p[0]=(bf16_t)f0.x; p[1]=(bf16_t)f0.y; p[2]=(bf16_t)f0.z; p[3]=(bf16_t)f0.w;
        p[4]=(bf16_t)f1.x; p[5]=(bf16_t)f1.y; p[6]=(bf16_t)f1.z; p[7]=(bf16_t)f1.w;
        *reinterpret_cast<v8bf*>(dst + 8 * j) = p;
      }
    }
    __syncthreads();

    // A fragment: convert 16 f32 of this lane's row on the fly
    v16bf a;
    {
      const float4 f0 = *reinterpret_cast<const float4*>(xrow + k0 + 8 * half);
      const float4 f1 = *reinterpret_cast<const float4*>(xrow + k0 + 8 * half + 4);
      const float4 f2 = *reinterpret_cast<const float4*>(xrow + k0 + 16 + 8 * half);
      const float4 f3 = *reinterpret_cast<const float4*>(xrow + k0 + 20 + 8 * half);
      a[0]=(bf16_t)f0.x; a[1]=(bf16_t)f0.y; a[2]=(bf16_t)f0.z; a[3]=(bf16_t)f0.w;
      a[4]=(bf16_t)f1.x; a[5]=(bf16_t)f1.y; a[6]=(bf16_t)f1.z; a[7]=(bf16_t)f1.w;
      a[8]=(bf16_t)f2.x; a[9]=(bf16_t)f2.y; a[10]=(bf16_t)f2.z; a[11]=(bf16_t)f2.w;
      a[12]=(bf16_t)f3.x; a[13]=(bf16_t)f3.y; a[14]=(bf16_t)f3.z; a[15]=(bf16_t)f3.w;
    }

#pragma unroll
    for (int t = 0; t < 16; ++t) {
      v16bf b = b_frag_bf16(ldsW + (16 * t + ln) * 32, 0, half);
      acc[t] = wmma_bf16(a, b, acc[t]);
    }
  }

  // ---- stores. Global row r = b*2048 + c.
  const int b_idx = r0 >> 11;
  const int cbase = (r0 & 2047) + 16 * wave;
#pragma unroll
  for (int t = 0; t < 16; ++t) {
    const int n = 16 * t + ln;
    if (Yrow) {  // row-major [B*C][256]
#pragma unroll
      for (int e = 0; e < 8; ++e) {
        const int r = r0 + 16 * wave + e + 8 * half;
        Yrow[(size_t)r * 256 + n] = (bf16_t)acc[t][e];
      }
    }
    if (Ytr) {   // transposed [B][256][2048]; rows of D tile are contiguous in c
      v8bf p;
#pragma unroll
      for (int e = 0; e < 8; ++e) p[e] = (bf16_t)acc[t][e];
      *reinterpret_cast<v8bf*>(Ytr + ((size_t)(b_idx * 256 + n)) * 2048 +
                               cbase + 8 * half) = p;
    }
  }
}

// ---------------------------------------------------------------------------
// Kernel 2: hstate transpose+convert: HT[b][d][h] = (bf16) hstate[b][h][d]
// ---------------------------------------------------------------------------
__global__ __launch_bounds__(256) void ht_kernel(const float* __restrict__ H,
                                                 bf16_t* __restrict__ HT) {
  const size_t i = (size_t)blockIdx.x * 256 + threadIdx.x;
  if (i >= (size_t)16 * 256 * 256) return;
  const int b = (int)(i >> 16), h = (int)((i >> 8) & 255), d = (int)(i & 255);
  HT[((size_t)b << 16) + (size_t)d * 256 + h] = (bf16_t)H[i];
}

// ---------------------------------------------------------------------------
// Kernel 3: retention core.  block = (batch, 128-row tile of C).
//   O = q @ HT^T  (cross chunk)
//   for causal/banded j-blocks: S = (q k^T) * kappa^(i-j);  O += S @ v
// S tiles go D-fragment -> LDS (bf16) -> A-fragment within the same wave.
// out1[b][i][d] = O  (f32)
// ---------------------------------------------------------------------------
__global__ __launch_bounds__(256) void ret_kernel(
    const bf16_t* __restrict__ q, const bf16_t* __restrict__ k,
    const bf16_t* __restrict__ vT, const bf16_t* __restrict__ HT,
    float* __restrict__ out1) {
  __shared__ bf16_t ldsS[8 * 16 * 32];   // 1 KB per wave

  const int tid  = threadIdx.x;
  const int wave = tid >> 5, lane = tid & 31;
  const int half = lane >> 4, ln = lane & 15;
  const int b    = blockIdx.y;
  const int i0   = blockIdx.x * 128;
  const int ibase = i0 + 16 * wave;

  const bf16_t* qb  = q  + (size_t)b * 2048 * 256;
  const bf16_t* kb  = k  + (size_t)b * 2048 * 256;
  const bf16_t* vTb = vT + (size_t)b * 256 * 2048;
  const bf16_t* HTb = HT + ((size_t)b << 16);
  const bf16_t* qrow = qb + (size_t)(ibase + ln) * 256;

  // Cache this wave's 8 q fragments (K=256)
  v16bf qa[8];
#pragma unroll
  for (int kk = 0; kk < 8; ++kk) qa[kk] = a_frag_bf16(qrow, kk * 32, half);

  v8f o[16];
#pragma unroll
  for (int t = 0; t < 16; ++t) o[t] = (v8f){0.f,0.f,0.f,0.f,0.f,0.f,0.f,0.f};

  // ---- cross chunk: O = q @ hstate  (B^T row-major = HT)
#pragma unroll
  for (int kk = 0; kk < 8; ++kk) {
#pragma unroll
    for (int t = 0; t < 16; ++t) {
      v16bf bf = b_frag_bf16(HTb + (size_t)(16 * t + ln) * 256, kk * 32, half);
      o[t] = wmma_bf16(qa[kk], bf, o[t]);
    }
  }

  // ---- banded causal loop. kappa^d underflows f32 for d ~> 1000, exactly as
  // the reference's f32 decay matrix does, so skip fully-zero j blocks.
  const int jstart = (i0 > 1056) ? ((i0 - 1056) & ~31) : 0;
  bf16_t* myS = ldsS + wave * 512;

  for (int j0 = jstart; j0 < i0 + 128; j0 += 32) {
    // S tile (16 rows x 32 cols) = q_i . k_j^T   (B^T row-major = k row-major)
    v8f s[2];
    s[0] = (v8f){0.f,0.f,0.f,0.f,0.f,0.f,0.f,0.f};
    s[1] = (v8f){0.f,0.f,0.f,0.f,0.f,0.f,0.f,0.f};
#pragma unroll
    for (int kk = 0; kk < 8; ++kk) {
#pragma unroll
      for (int t2 = 0; t2 < 2; ++t2) {
        v16bf bf = b_frag_bf16(kb + (size_t)(j0 + 16 * t2 + ln) * 256, kk * 32, half);
        s[t2] = wmma_bf16(qa[kk], bf, s[t2]);
      }
    }
    // decay + causal mask, convert to bf16, stage into per-wave LDS (row-major 16x32)
#pragma unroll
    for (int t2 = 0; t2 < 2; ++t2) {
#pragma unroll
      for (int e = 0; e < 8; ++e) {
        const int i = ibase + e + 8 * half;
        const int j = j0 + 16 * t2 + ln;
        const int d = i - j;
        const float dec = (d >= 0) ? __expf(LN_KAPPA * (float)d) : 0.0f;
        myS[(e + 8 * half) * 32 + 16 * t2 + ln] = (bf16_t)(s[t2][e] * dec);
      }
    }
    // Same-wave DS ops are in-order; make the RAW explicit.
    asm volatile("s_wait_dscnt 0" ::: "memory");

    // Reload S as an A fragment, then O += S @ v  (B^T row-major = vT)
    v16bf sa = a_frag_bf16(myS + ln * 32, 0, half);
#pragma unroll
    for (int t = 0; t < 16; ++t) {
      v16bf vb = b_frag_bf16(vTb + (size_t)(16 * t + ln) * 2048, j0, half);
      o[t] = wmma_bf16(sa, vb, o[t]);
    }
  }

  // ---- store O (inner + cross), f32
  float* outb = out1 + (size_t)b * 2048 * 256;
#pragma unroll
  for (int t = 0; t < 16; ++t) {
    const int n = 16 * t + ln;
#pragma unroll
    for (int e = 0; e < 8; ++e) {
      const int i = ibase + e + 8 * half;
      outb[(size_t)i * 256 + n] = o[t][e];
    }
  }
}

// ---------------------------------------------------------------------------
// Kernel 4: next_hstate = kT(256x2048) @ (v * lastdecay)(2048x256)
//                         + hstate * kappa^256 * delta
// lastdecay[c] = kappa^(2047-c) underflows f32 for c < ~1060 -> start at 1024.
// block = (half of 256 rows, batch). delta from dones[b, ::8] reduction.
// ---------------------------------------------------------------------------
__global__ __launch_bounds__(256) void hstate_kernel(
    const bf16_t* __restrict__ kT, const bf16_t* __restrict__ vT,
    const float* __restrict__ hstate, const unsigned char* __restrict__ dones,
    float* __restrict__ out2) {
  __shared__ int flag;
  const int tid  = threadIdx.x;
  const int wave = tid >> 5, lane = tid & 31;
  const int half = lane >> 4, ln = lane & 15;
  const int b    = blockIdx.y;

  if (tid == 0) flag = 0;
  __syncthreads();
  if (dones[(size_t)b * 2048 + 8 * tid]) atomicOr(&flag, 1);
  __syncthreads();
  const float delta = flag ? 0.0f : 1.0f;

  const int mbase = blockIdx.x * 128 + 16 * wave;
  const bf16_t* arow = kT + (size_t)b * 256 * 2048 + (size_t)(mbase + ln) * 2048;
  const bf16_t* vTb  = vT + (size_t)b * 256 * 2048;

  v8f o[16];
#pragma unroll
  for (int t = 0; t < 16; ++t) o[t] = (v8f){0.f,0.f,0.f,0.f,0.f,0.f,0.f,0.f};

  const float invK = 1.0f / 0.9f;
  for (int c0 = 1024; c0 < 2048; c0 += 32) {
    v16bf a = a_frag_bf16(arow, c0, half);
    // lastdecay for the 16 K-slots of this lane's B fragment: c = c0+e+8*half (e<8), c0+e+8+8*half (e>=8)
    float ldv[16];
    float m = __expf(LN_KAPPA * (float)(2047 - c0 - 8 * half));
#pragma unroll
    for (int e = 0; e < 16; ++e) { if (e == 8) m = __expf(LN_KAPPA * (float)(2047 - c0 - 16 - 8 * half)); ldv[e] = m; m *= invK; }

#pragma unroll
    for (int t = 0; t < 16; ++t) {
      v16bf raw = b_frag_bf16(vTb + (size_t)(16 * t + ln) * 2048, c0, half);
      v16bf vb;
#pragma unroll
      for (int e = 0; e < 16; ++e) vb[e] = (bf16_t)((float)raw[e] * ldv[e]);
      o[t] = wmma_bf16(a, vb, o[t]);
    }
  }

  const float cd = __expf(LN_KAPPA * 256.0f) * delta;  // kappa^256 * delta
  float* outp = out2 + ((size_t)b << 16);
  const float* hb = hstate + ((size_t)b << 16);
#pragma unroll
  for (int t = 0; t < 16; ++t) {
    const int n = 16 * t + ln;
#pragma unroll
    for (int e = 0; e < 8; ++e) {
      const int mm = mbase + e + 8 * half;
      outp[(size_t)mm * 256 + n] = o[t][e] + hb[(size_t)mm * 256 + n] * cd;
    }
  }
}

// ---------------------------------------------------------------------------
// kernel_launch
// input order: key, query, value, hstate, dones, w_q, w_k, w_v
// output: [inner+cross (16,2048,256) f32][next_hstate (16,256,256) f32]
// workspace: q,k,kT,vT bf16 (16 MB each) + HT bf16 (2 MB) = 66 MB
// ---------------------------------------------------------------------------
extern "C" void kernel_launch(void* const* d_in, const int* in_sizes, int n_in,
                              void* d_out, int out_size, void* d_ws, size_t ws_size,
                              hipStream_t stream) {
  const float* key    = (const float*)d_in[0];
  const float* query  = (const float*)d_in[1];
  const float* value  = (const float*)d_in[2];
  const float* hstate = (const float*)d_in[3];
  const unsigned char* dones = (const unsigned char*)d_in[4];
  const float* w_q = (const float*)d_in[5];
  const float* w_k = (const float*)d_in[6];
  const float* w_v = (const float*)d_in[7];

  const size_t SZ = (size_t)16 * 2048 * 256;   // 8388608
  bf16_t* q  = (bf16_t*)d_ws;
  bf16_t* k  = q  + SZ;
  bf16_t* kT = k  + SZ;
  bf16_t* vT = kT + SZ;
  bf16_t* HT = vT + SZ;                        // 16*256*256

  float* out1 = (float*)d_out;
  float* out2 = out1 + SZ;

  dim3 blk(256);
  proj_kernel<<<dim3(256), blk, 0, stream>>>(query, w_q, q, (bf16_t*)nullptr);
  proj_kernel<<<dim3(256), blk, 0, stream>>>(key,   w_k, k, kT);
  proj_kernel<<<dim3(256), blk, 0, stream>>>(value, w_v, (bf16_t*)nullptr, vT);
  ht_kernel<<<dim3(4096), blk, 0, stream>>>(hstate, HT);
  ret_kernel<<<dim3(16, 16), blk, 0, stream>>>(q, k, vT, HT, out1);
  hstate_kernel<<<dim3(2, 16), blk, 0, stream>>>(kT, vT, hstate, dones, out2);
}
